// CompressedInteractionNetwork_5506148073814
// MI455X (gfx1250) — matrix operand order repaired and checked by reading, not compiled, not run on hardware
//
#include <hip/hip_runtime.h>
#include <hip/hip_bf16.h>

// ---------------------------------------------------------------------------
// CIN (xDeepFM Compressed Interaction Network) forward on gfx1250.
// B=4096, F=32, E=64, CROSS=[256,256,256].  GEMMs run on
// v_wmma_f32_16x16x32_bf16 with fp32 accumulation; Z (outer-product feature
// map) fragments are built in registers with packed bf16 multiplies
// (co-executes with XDL).  Wave tiling: 8 waves = 8 o-groups; each wave owns
// 2 o-tiles x 4 e-tiles (32 rows x all 64 cols) so every weight row is read
// by exactly one wave per block (halves L2 A-traffic vs 4x2 tiling).
// K-loop is software-pipelined (step 64, double-buffered A/B fragments).
// ---------------------------------------------------------------------------

typedef __attribute__((ext_vector_type(16))) __bf16 v16bf;
typedef __attribute__((ext_vector_type(8)))  __bf16 v8bf;
typedef __attribute__((ext_vector_type(8)))  float  v8f;

#define CIN_B   4096
#define CIN_F   32
#define CIN_E   64
#define CIN_O   256        // rows per cross layer
#define C0      1024       // F*F
#define C12     4096       // F*128
#define W0_ELEMS   (CIN_O * C0)                 // 262144
#define W12_ELEMS  (CIN_O * C12)                // 1048576
#define W_TOTAL    (W0_ELEMS + 2 * W12_ELEMS)   // 2359296 bf16 elements
#define HTS     136        // padded row stride (bf16) of transposed h buffers
#define XTS     40         // padded row stride (bf16) of transposed x buffer

// ---------------------------------------------------------------------------
// Kernel 1: convert fp32 weights -> contiguous bf16 copy in workspace.
// Layout: [w0 (256x1024) | w1 (256x4096) | w2 (256x4096)], row-major.
// ---------------------------------------------------------------------------
__global__ void cin_convert_weights(const float* __restrict__ w0,
                                    const float* __restrict__ w1,
                                    const float* __restrict__ w2,
                                    __bf16* __restrict__ wsbf) {
    unsigned i = blockIdx.x * blockDim.x + threadIdx.x;
    if (i >= W_TOTAL) return;
    float v;
    if (i < W0_ELEMS)                  v = w0[i];
    else if (i < W0_ELEMS + W12_ELEMS) v = w1[i - W0_ELEMS];
    else                               v = w2[i - (W0_ELEMS + W12_ELEMS)];
    wsbf[i] = (__bf16)v;
}

// raw (unscaled) B-matrix source data for one 32-deep K step (4 e-tiles)
struct BSrc {
    v8bf  h0[4], h1[4];   // 16 consecutive h values per e-tile
    __bf16 xv[4];         // broadcast x scale per e-tile
};

// ---------------------------------------------------------------------------
// Kernel 2: one workgroup per batch element, 8 waves (wave32).
// ---------------------------------------------------------------------------
__global__ __launch_bounds__(256)
void cin_forward(const float* __restrict__ x,       // (B, 32, 64)
                 const __bf16* __restrict__ wsbf,   // packed bf16 weights
                 const float* __restrict__ b0,
                 const float* __restrict__ b1,
                 const float* __restrict__ b2,
                 const float* __restrict__ fcw,     // (1, 512)
                 const float* __restrict__ fcb,     // (1,)
                 float* __restrict__ out) {         // (B, 1)
    __shared__ __bf16 xT [CIN_E * XTS];        // x_b transposed bf16 (5 KB)
    __shared__ __bf16 hT0[CIN_E * HTS];        // h (z0[128:]) transposed (17 KB)
    __shared__ __bf16 hT1[CIN_E * HTS];        // h (z1[128:]) transposed (17 KB)
    __shared__ float  biasL[3 * CIN_O];        // all three biases (3 KB)
    __shared__ float  fcl[512];                // fc weights (2 KB)
    __shared__ float  red[256];                // reduction scratch

    const int tid  = threadIdx.x;
    const int b    = blockIdx.x;
    const int lane = tid & 31;
    const int wave = tid >> 5;
    const int n    = lane & 15;     // column within 16-wide tile
    const int sel  = lane >> 4;     // which half of the wave
    const int oBase = wave * 32;    // this wave's 32-row output stripe

    // ---- stage x_b (transposed bf16), biases, fc weights ----
    const float* xb = x + (size_t)b * (CIN_F * CIN_E);
    for (int i = tid; i < CIN_F * CIN_E; i += 256) {
        int f = i >> 6, e = i & 63;
        xT[e * XTS + f] = (__bf16)xb[i];
    }
    for (int i = tid; i < 3 * CIN_O; i += 256) {
        float v = (i < 256) ? b0[i] : (i < 512 ? b1[i - 256] : b2[i - 512]);
        biasL[i] = v;
    }
    for (int i = tid; i < 512; i += 256) fcl[i] = fcw[i];
    __syncthreads();

    // ---- one cross layer: W(256xC) x Z(Cx64), bias, relu, pool, emit h ----
    auto layer = [&](const __bf16* __restrict__ Wb, int C, int shift,
                     const __bf16* __restrict__ hT, int hts,
                     int biasOff, int fcBase, int keep,
                     __bf16* __restrict__ hTout) -> float {
        v8f acc[2][4] = {};
        const int mask = (1 << shift) - 1;

        // A fragments: two 16x32 bf16 weight tiles (contiguous striping:
        // lanes 0-15 hold K 0-15, lanes 16-31 hold K 16-31)
        auto loadA = [&](int k, v16bf (&a)[2]) {
#pragma unroll
            for (int t = 0; t < 2; ++t) {
                const __bf16* wr =
                    Wb + (size_t)(oBase + t * 16 + n) * C + k + sel * 16;
                v8bf lo = *(const v8bf*)wr;
                v8bf hi = *(const v8bf*)(wr + 8);
                a[t] = __builtin_shufflevector(
                    lo, hi, 0, 1, 2, 3, 4, 5, 6, 7,
                    8, 9, 10, 11, 12, 13, 14, 15);
                if (k + 64 < C) __builtin_prefetch(wr + 64, 0, 1);
            }
        };
        // raw B source: h vectors + x broadcast scalar for one K step
        auto loadB = [&](int k) -> BSrc {
            BSrc s;
            const int cBase = k + sel * 16;
            const int cHi   = cBase >> shift;
            const int cLo   = cBase & mask;
#pragma unroll
            for (int et = 0; et < 4; ++et) {
                const int col = et * 16 + n;
                s.xv[et] = xT[col * XTS + cHi];
                const v8bf* hp = (const v8bf*)(hT + col * hts + cLo);
                s.h0[et] = hp[0];
                s.h1[et] = hp[1];
            }
            return s;
        };
        // scale h by x (packed bf16, co-executes with XDL) and run 8 WMMAs
        auto compute = [&](const v16bf (&a)[2], const BSrc& s) {
#pragma unroll
            for (int et = 0; et < 4; ++et) {
                v8bf xs;
#pragma unroll
                for (int j = 0; j < 8; ++j) xs[j] = s.xv[et];
                v8bf p0 = s.h0[et] * xs;   // v_pk_mul_bf16 x4
                v8bf p1 = s.h1[et] * xs;   // v_pk_mul_bf16 x4
                v16bf bfrag = __builtin_shufflevector(
                    p0, p1, 0, 1, 2, 3, 4, 5, 6, 7,
                    8, 9, 10, 11, 12, 13, 14, 15);
#pragma unroll
                for (int t = 0; t < 2; ++t)
                    acc[t][et] = __builtin_amdgcn_wmma_f32_16x16x32_bf16(
                        false, a[t], false, bfrag, (short)0, acc[t][et],
                        false, false);
            }
        };

        // software-pipelined K loop: loads run one 32-step ahead of compute
        v16bf aA[2], aB[2];
        loadA(0, aA);
        BSrc sA = loadB(0);
        for (int k = 0; k < C; k += 64) {
            loadA(k + 32, aB);
            BSrc sB = loadB(k + 32);
            compute(aA, sA);
            if (k + 64 < C) {
                loadA(k + 64, aA);
                sA = loadB(k + 64);
            }
            compute(aB, sB);
        }

        // epilogue: bias + relu, pooled fc contribution, transposed h write
        float pool = 0.f;
#pragma unroll
        for (int t = 0; t < 2; ++t) {
            const int oT = oBase + t * 16;
#pragma unroll
            for (int et = 0; et < 4; ++et) {
                const int e = et * 16 + n;
                v8bf pack;
#pragma unroll
                for (int r = 0; r < 8; ++r) {
                    const int o = oT + sel * 8 + r;   // C/D VGPR layout
                    float v = acc[t][et][r] + biasL[biasOff + o];
                    v = fmaxf(v, 0.f);
                    if (o < keep) pool += fcl[fcBase + o] * v;
                    pack[r] = (__bf16)v;
                }
                if (hTout && oT >= 128)
                    *(v8bf*)(hTout + e * HTS + (oT - 128 + sel * 8)) = pack;
            }
        }
        return pool;
    };

    float pool = 0.f;
    // layer 0: C=1024, h = x itself (transposed bf16, stride XTS)
    pool += layer(wsbf, C0, 5, xT, XTS, /*bias*/0, /*fc*/0, /*keep*/128, hT0);
    __syncthreads();
    // layer 1: C=4096, h = z0[128:256] (hT0)
    pool += layer(wsbf + W0_ELEMS, C12, 7, hT0, HTS, 256, 128, 128, hT1);
    __syncthreads();
    // layer 2: C=4096, h = z1[128:256] (hT1); keep all 256 rows, no h out
    pool += layer(wsbf + W0_ELEMS + W12_ELEMS, C12, 7, hT1, HTS,
                  512, 256, 256, (__bf16*)nullptr);

    // ---- block reduction of pooled dot-product ----
    red[tid] = pool;
    __syncthreads();
#pragma unroll
    for (int s = 128; s > 0; s >>= 1) {
        if (tid < s) red[tid] += red[tid + s];
        __syncthreads();
    }
    if (tid == 0) out[b] = red[0] + fcb[0];
}

// ---------------------------------------------------------------------------
// Host launcher
// ---------------------------------------------------------------------------
extern "C" void kernel_launch(void* const* d_in, const int* in_sizes, int n_in,
                              void* d_out, int out_size, void* d_ws,
                              size_t ws_size, hipStream_t stream) {
    (void)in_sizes; (void)n_in; (void)out_size;
    const float* x   = (const float*)d_in[0];
    const float* w0  = (const float*)d_in[1];
    const float* b0  = (const float*)d_in[2];
    const float* w1  = (const float*)d_in[3];
    const float* b1  = (const float*)d_in[4];
    const float* w2  = (const float*)d_in[5];
    const float* b2  = (const float*)d_in[6];
    const float* fcw = (const float*)d_in[7];
    const float* fcb = (const float*)d_in[8];
    float* out       = (float*)d_out;

    if (ws_size < (size_t)W_TOTAL * sizeof(__bf16)) return;
    __bf16* wsbf = (__bf16*)d_ws;

    cin_convert_weights<<<(W_TOTAL + 255) / 256, 256, 0, stream>>>(w0, w1, w2,
                                                                   wsbf);
    cin_forward<<<CIN_B, 256, 0, stream>>>(x, wsbf, b0, b1, b2, fcw, fcb, out);
}
